// MultiHeadAttention_90795608637762
// MI455X (gfx1250) — compile-verified, matrix operand order checked by reading, not run on hardware
//
#include <hip/hip_runtime.h>

typedef __bf16 bf16_t;
typedef __attribute__((ext_vector_type(16))) __bf16 v16bf;
typedef __attribute__((ext_vector_type(8)))  __bf16 v8bf;
typedef __attribute__((ext_vector_type(8)))  float  v8f;

#define DEVI __device__ __forceinline__

constexpr int S_TOK  = 2048;
constexpr int DMODEL = 2048;
constexpr int NHEADS = 16;
constexpr int HDIM   = 128;
constexpr float ATT_SCALE = 0.08838834764831845f;   // 1/sqrt(128)
constexpr float NEG_BIG   = -3.0e38f;
constexpr float ROPE_LN   = 0.14391156509613672f;   // ln(10000)/64

DEVI v16bf cat8(v8bf lo, v8bf hi) {
  return __builtin_shufflevector(lo, hi, 0,1,2,3,4,5,6,7,8,9,10,11,12,13,14,15);
}

DEVI v16bf frag_ld(const bf16_t* base) {
  v8bf lo = *(const v8bf*)base;
  v8bf hi = *(const v8bf*)(base + 16);
  return cat8(lo, hi);
}

// A/B fragment load from a row-major bf16 tile (B tiles are staged transposed
// so both use the 16-bit 16xK WMMA operand layout: lane = row, two 16B chunks).
DEVI v16bf frag_ld_rm(const bf16_t* p, int stride, int row0, int col0, int lane) {
  const bf16_t* base = p + (row0 + (lane & 15)) * stride + col0 + 8 * (lane >> 4);
  return frag_ld(base);
}

DEVI v8f wmma_bf16(v16bf a, v16bf b, v8f c) {
  return __builtin_amdgcn_wmma_f32_16x16x32_bf16(false, a, false, b, (short)0, c,
                                                 false, false);
}

// gfx1250 async DMA: copy 16 bytes/lane global -> LDS, tracked on ASYNCcnt.
DEVI void async_g2l_b128(unsigned lds_off, const bf16_t* gaddr) {
  asm volatile("global_load_async_to_lds_b128 %0, %1, off"
               :: "v"(lds_off), "v"(gaddr) : "memory");
}

// ---------------------------------------------------------------------------
// Generic 128x128 tile GEMM: C = A[M,K] @ W[K,N] + bias, with epilogues.
// MODE 0: Q  -> RoPE, bf16 store to [B,H,S,hd]
// MODE 1: K  -> RoPE, bf16 ws store + fp32 store (kh)
// MODE 2: V  -> bf16 ws store + fp32 store (vh)
// MODE 3: O  -> fp32 store to out
// ---------------------------------------------------------------------------
template <int MODE, bool A_BF16>
__global__ __launch_bounds__(256) void gemm_bias_rope(
    const void* __restrict__ Aptr, const float* __restrict__ W,
    const float* __restrict__ bias, bf16_t* __restrict__ out_bf,
    float* __restrict__ out_f, int K, int N) {
  __shared__ bf16_t ldsA[128 * 40];   // A tile, row-major [m][k]
  __shared__ bf16_t ldsB[128 * 40];   // W tile, transposed [n][k]
  const int tid  = threadIdx.x;
  const int lane = tid & 31;
  const int wave = tid >> 5;
  const int wm   = wave >> 1;         // 4 wave-rows of 32
  const int wn   = wave & 1;          // 2 wave-cols of 64
  const int m0   = blockIdx.x * 128;
  const int n0   = blockIdx.y * 128;

  v8f acc[2][4] = {};

  for (int k0 = 0; k0 < K; k0 += 32) {
    #pragma unroll
    for (int i = 0; i < 16; ++i) {           // stage A 128x32 -> bf16
      int idx = i * 256 + tid;
      int r = idx >> 5, c = idx & 31;
      float v;
      if (A_BF16)
        v = (float)((const bf16_t*)Aptr)[(size_t)(m0 + r) * K + k0 + c];
      else
        v = ((const float*)Aptr)[(size_t)(m0 + r) * K + k0 + c];
      ldsA[r * 40 + c] = (bf16_t)v;
    }
    #pragma unroll
    for (int i = 0; i < 16; ++i) {           // stage W 32x128 -> transposed
      int idx = i * 256 + tid;
      int r = idx >> 7, c = idx & 127;
      ldsB[c * 40 + r] = (bf16_t)W[(size_t)(k0 + r) * N + n0 + c];
    }
    if (k0 + 32 < K)                          // global_prefetch_b8
      __builtin_prefetch((const char*)W + sizeof(float) * ((size_t)(k0 + 32) * N + n0), 0, 1);
    __syncthreads();

    v16bf afr[2], bfr[4];
    #pragma unroll
    for (int mf = 0; mf < 2; ++mf) afr[mf] = frag_ld_rm(ldsA, 40, wm * 32 + mf * 16, 0, lane);
    #pragma unroll
    for (int nf = 0; nf < 4; ++nf) bfr[nf] = frag_ld_rm(ldsB, 40, wn * 64 + nf * 16, 0, lane);
    #pragma unroll
    for (int mf = 0; mf < 2; ++mf)
      #pragma unroll
      for (int nf = 0; nf < 4; ++nf)
        acc[mf][nf] = wmma_bf16(afr[mf], bfr[nf], acc[mf][nf]);
    __syncthreads();
  }

  // Epilogue. C layout: VGPR r, lanes0-15: (M=r, N=lane); lanes16-31: (M=r+8).
  #pragma unroll
  for (int mf = 0; mf < 2; ++mf) {
    #pragma unroll
    for (int nf = 0; nf < 4; ++nf) {
      const int n  = n0 + wn * 64 + nf * 16 + (lane & 15);
      const float bv = bias[n];
      #pragma unroll
      for (int r = 0; r < 8; ++r) {
        const int m = m0 + wm * 32 + mf * 16 + r + 8 * (lane >> 4);
        float x = acc[mf][nf][r] + bv;
        float o = x;
        if constexpr (MODE == 0 || MODE == 1) {  // RoPE: pair partner on lane^1
          const int srow = m & (S_TOK - 1);
          const int j = (n & (HDIM - 1)) >> 1;
          const float ang = (float)srow * __expf(-ROPE_LN * (float)j);
          float sn, cs;
          __sincosf(ang, &sn, &cs);
          const float partner = __shfl_xor(x, 1, 32);
          o = (n & 1) ? (partner * sn + x * cs) : (x * cs - partner * sn);
        }
        if constexpr (MODE == 0) {
          const int b = m >> 11, srow = m & (S_TOK - 1);
          const int h = n >> 7, d = n & (HDIM - 1);
          out_bf[((size_t)(b * NHEADS + h) * S_TOK + srow) * HDIM + d] = (bf16_t)o;
        } else if constexpr (MODE == 1 || MODE == 2) {
          const size_t idx = (size_t)m * HDIM + n;
          out_bf[idx] = (bf16_t)o;
          out_f[idx]  = o;
        } else {
          out_f[(size_t)m * DMODEL + n] = o;
        }
      }
    }
  }
}

// ---------------------------------------------------------------------------
// Flash-style causal MQA attention. Block = 4 waves x 16 query rows.
// K tile staged via gfx1250 async global->LDS DMA (ASYNCcnt path).
// ---------------------------------------------------------------------------
__global__ __launch_bounds__(128) void attn_mqa(
    const bf16_t* __restrict__ qh, const bf16_t* __restrict__ kws,
    const bf16_t* __restrict__ vws, bf16_t* __restrict__ aout) {
  __shared__ bf16_t ldsK[32 * 136];       // [key][hd]  (== B-layout for Q@K^T)
  __shared__ bf16_t ldsV[128 * 40];       // [hd][key]  (transposed V, B-layout)
  __shared__ bf16_t ldsP[4][16 * 32];     // per-wave P tile

  const int tid  = threadIdx.x;
  const int lane = tid & 31;
  const int wave = tid >> 5;
  const int hi   = lane >> 4;
  const int b = blockIdx.z, h = blockIdx.y;
  const int q0 = blockIdx.x * 64;
  const int qrow = q0 + wave * 16;

  // resident Q fragments: 16 rows x 128 (K=4 steps of 32)
  const bf16_t* qbase = qh + ((size_t)(b * NHEADS + h) * S_TOK + qrow) * HDIM;
  v16bf qf[4];
  #pragma unroll
  for (int kk = 0; kk < 4; ++kk)
    qf[kk] = frag_ld(qbase + (lane & 15) * HDIM + kk * 32 + 8 * hi);

  v8f o[8] = {};
  float mst[8], lst[8];
  #pragma unroll
  for (int r = 0; r < 8; ++r) { mst[r] = NEG_BIG; lst[r] = 0.f; }

  const int kend = q0 + 64;
  for (int k0 = 0; k0 < kend; k0 += 32) {
    // K tile: straight byte copy -> async DMA, 16B per lane per issue
    #pragma unroll
    for (int i = 0; i < 4; ++i) {
      int e = (i * 128 + tid) * 8;          // element offset in 32x128 tile
      int r = e >> 7, c = e & 127;
      unsigned lds_off = (unsigned)(size_t)&ldsK[r * 136 + c];
      async_g2l_b128(lds_off, kws + ((size_t)b * S_TOK + k0 + r) * HDIM + c);
    }
    // V tile staged transposed (data must pass through lanes)
    #pragma unroll
    for (int i = 0; i < 32; ++i) {
      int idx = i * 128 + tid;
      int r = idx >> 7, c = idx & 127;
      ldsV[c * 40 + r] = vws[((size_t)b * S_TOK + k0 + r) * HDIM + c];
    }
    asm volatile("s_wait_asynccnt 0" ::: "memory");
    __syncthreads();

    v8f sf[2] = {};
    #pragma unroll
    for (int ft = 0; ft < 2; ++ft)
      #pragma unroll
      for (int kk = 0; kk < 4; ++kk) {
        v16bf kf = frag_ld_rm(ldsK, 136, ft * 16, kk * 32, lane);
        sf[ft] = wmma_bf16(qf[kk], kf, sf[ft]);
      }

    // scale + causal mask + online softmax (row = 16-lane half reductions)
    const int keycol = k0 + (lane & 15);
    float oscale[8];
    #pragma unroll
    for (int r = 0; r < 8; ++r) {
      const int qr = qrow + r + 8 * hi;
      float v0 = sf[0][r] * ATT_SCALE;
      float v1 = sf[1][r] * ATT_SCALE;
      if (keycol > qr)      v0 = NEG_BIG;
      if (keycol + 16 > qr) v1 = NEG_BIG;
      float vmax = fmaxf(v0, v1);
      #pragma unroll
      for (int off = 1; off < 16; off <<= 1)
        vmax = fmaxf(vmax, __shfl_xor(vmax, off, 32));
      const float mnew = fmaxf(mst[r], vmax);
      const float scl  = __expf(mst[r] - mnew);
      const float p0 = __expf(v0 - mnew);
      const float p1 = __expf(v1 - mnew);
      float ps = p0 + p1;
      #pragma unroll
      for (int off = 1; off < 16; off <<= 1)
        ps += __shfl_xor(ps, off, 32);
      lst[r] = lst[r] * scl + ps;
      mst[r] = mnew;
      oscale[r] = scl;
      bf16_t* pr = &ldsP[wave][(r + 8 * hi) * 32 + (lane & 15)];
      pr[0]  = (bf16_t)p0;
      pr[16] = (bf16_t)p1;
    }
    asm volatile("s_wait_dscnt 0" ::: "memory");

    #pragma unroll
    for (int nf = 0; nf < 8; ++nf)
      #pragma unroll
      for (int r = 0; r < 8; ++r) o[nf][r] *= oscale[r];

    v16bf pf = frag_ld_rm(ldsP[wave], 32, 0, 0, lane);
    #pragma unroll
    for (int nf = 0; nf < 8; ++nf) {
      v16bf vf = frag_ld_rm(ldsV, 40, nf * 16, 0, lane);
      o[nf] = wmma_bf16(pf, vf, o[nf]);
    }
    __syncthreads();
  }

  // normalize + store to attention workspace [B*S, D] bf16
  #pragma unroll
  for (int nf = 0; nf < 8; ++nf)
    #pragma unroll
    for (int r = 0; r < 8; ++r) {
      const int qr = qrow + r + 8 * hi;
      const float v = o[nf][r] / lst[r];
      aout[((size_t)b * S_TOK + qr) * DMODEL + h * HDIM + nf * 16 + (lane & 15)] =
          (bf16_t)v;
    }
}

extern "C" void kernel_launch(void* const* d_in, const int* in_sizes, int n_in,
                              void* d_out, int out_size, void* d_ws, size_t ws_size,
                              hipStream_t stream) {
  const float* x  = (const float*)d_in[0];
  const float* wq = (const float*)d_in[1];
  const float* bq = (const float*)d_in[2];
  const float* wk = (const float*)d_in[3];
  const float* bk = (const float*)d_in[4];
  const float* wv = (const float*)d_in[5];
  const float* bv = (const float*)d_in[6];
  const float* wo = (const float*)d_in[7];
  const float* bo = (const float*)d_in[8];

  float* out = (float*)d_out;                        // [4,2048,2048]
  float* kh  = out + (size_t)4 * 2048 * 2048;        // [4,1,2048,128]
  float* vh  = kh + (size_t)4 * 2048 * 128;          // [4,1,2048,128]

  bf16_t* q_bf = (bf16_t*)d_ws;                      // [B,H,S,hd]
  bf16_t* k_bf = q_bf + (size_t)4 * 2048 * 2048;
  bf16_t* v_bf = k_bf + (size_t)4 * 2048 * 128;
  bf16_t* a_bf = v_bf + (size_t)4 * 2048 * 128;      // [B*S, D]

  gemm_bias_rope<0, false><<<dim3(64, 16), 256, 0, stream>>>(x, wq, bq, q_bf, nullptr, 2048, 2048);
  gemm_bias_rope<1, false><<<dim3(64, 1),  256, 0, stream>>>(x, wk, bk, k_bf, kh,      2048, 128);
  gemm_bias_rope<2, false><<<dim3(64, 1),  256, 0, stream>>>(x, wv, bv, v_bf, vh,      2048, 128);
  attn_mqa<<<dim3(32, 16, 4), 128, 0, stream>>>(q_bf, k_bf, v_bf, a_bf);
  gemm_bias_rope<3, true><<<dim3(64, 16), 256, 0, stream>>>(a_bf, wo, bo, nullptr, out, 2048, 2048);
}
